// SeqAttentionModel_66168266162665
// MI455X (gfx1250) — compile-verified, matrix operand order checked by reading, not run on hardware
//
#include <hip/hip_runtime.h>

// ---------------------------------------------------------------------------
// MI455X (gfx1250, wave32) ragged-segment attention pooling.
//  - x@w1 (68.7 GFLOP) and post-FC GEMM on v_wmma_f32_16x16x32_bf16.
//  - w1^T panels staged to LDS with gfx1250 async copies
//    (global_load_async_to_lds_b128 + s_wait_asynccnt), double buffered, so
//    8 waves share one B-panel fetch and the DMA overlaps WMMA compute.
//  - A->H projection stays on the WMMA pipe via LDS relayout + padded w2.
// ---------------------------------------------------------------------------

typedef __attribute__((ext_vector_type(16))) __bf16 v16bf;
typedef __attribute__((ext_vector_type(8)))  float  v8f;

#define N_FR   16384
#define D_EMB  1024
#define A_ATT  2048
#define H_HEAD 8
#define B_SEG  64

#define BS_STRIDE 1032   // halves; 2064B rows -> bank-group skew of 4 per row

// ---- WMMA fragment helpers -------------------------------------------------
// 16-bit A-matrix (16xK tile, row-major source, row stride `stride` halves):
//   lane L: r = L&15, hi = L>>4 holds [row, k0+hi*8 .. +7] and
//   [row, k0+16+hi*8 .. +7] -> two contiguous 16-byte chunks.
// B-matrix (Kx16) layout is the mirror image with lane = column, so the same
// loader works on a column-major (pre-transposed) B source.
__device__ __forceinline__ v16bf frag_ld(const __bf16* base, int stride,
                                         int row, int k0, int hi) {
  const __bf16* q = base + (size_t)row * stride + k0 + hi * 8;
  v16bf f;
  reinterpret_cast<uint4*>(&f)[0] = *reinterpret_cast<const uint4*>(q);
  reinterpret_cast<uint4*>(&f)[1] = *reinterpret_cast<const uint4*>(q + 16);
  return f;
}

__device__ __forceinline__ v8f zero8() {
  v8f z;
#pragma unroll
  for (int i = 0; i < 8; ++i) z[i] = 0.0f;
  return z;
}

// Async-copy one 32-row x 1024-col bf16 panel of w1T into LDS (padded rows).
// 256 threads: thread -> row = tid>>3, 16 chunks of 16B spaced 8 chunks apart.
__device__ __forceinline__ void stage_panel_async(const __bf16* __restrict__ gsrc,
                                                  __bf16* lbuf, int tid) {
  const int rr = tid >> 3;          // 0..31 (a-column within panel)
  const int c0 = tid & 7;           // chunk phase 0..7
  const __bf16* g = gsrc + (size_t)rr * D_EMB;
  __bf16* l = lbuf + rr * BS_STRIDE;
#pragma unroll
  for (int j = 0; j < 16; ++j) {
    const int c = c0 + 8 * j;       // 0..127 (16B chunks along K)
    unsigned lds_addr = (unsigned)(size_t)(l + c * 8);
    unsigned long long gaddr = (unsigned long long)(const void*)(g + c * 8);
    asm volatile("global_load_async_to_lds_b128 %0, %1, off"
                 :: "v"(lds_addr), "v"(gaddr) : "memory");
  }
}

// ---- prep kernels ----------------------------------------------------------
__global__ void __launch_bounds__(64) k_offs(const int* __restrict__ seg,
                                             int* __restrict__ offs) {
  if (threadIdx.x == 0) {
    int a = 0;
    for (int b = 0; b < B_SEG; ++b) { offs[b] = a; a += seg[b]; }
    offs[B_SEG] = a;
  }
}

__global__ void __launch_bounds__(256) k_cvt_x(const float* __restrict__ x,
                                               __bf16* __restrict__ xb) {
  size_t i = (size_t)blockIdx.x * 256 + threadIdx.x;   // N*D elements
  xb[i] = (__bf16)x[i];
}

// w1 [D][A] f32 -> w1T [A][D] bf16 (column-major B source)
__global__ void __launch_bounds__(256) k_w1T(const float* __restrict__ w1,
                                             __bf16* __restrict__ w1T) {
  size_t i = (size_t)blockIdx.x * 256 + threadIdx.x;   // A*D elements
  int a = (int)(i >> 10);
  int k = (int)(i & 1023);
  w1T[i] = (__bf16)w1[(size_t)k * A_ATT + a];
}

// w2 [A][H] f32 -> w2T [16][A] bf16, columns 8..15 zero-padded
__global__ void __launch_bounds__(256) k_w2T(const float* __restrict__ w2,
                                             __bf16* __restrict__ w2T) {
  size_t i = (size_t)blockIdx.x * 256 + threadIdx.x;   // 16*A elements
  int c = (int)(i >> 11);
  int a = (int)(i & 2047);
  w2T[i] = (__bf16)((c < H_HEAD) ? w2[(size_t)a * H_HEAD + c] : 0.0f);
}

// w_post [D*H][D] f32 -> wpT [D][D*H] bf16 (column-major B source)
__global__ void __launch_bounds__(256) k_wpT(const float* __restrict__ wp,
                                             __bf16* __restrict__ wpT) {
  size_t i = (size_t)blockIdx.x * 256 + threadIdx.x;   // D * (D*H) elements
  int nn = (int)(i >> 13);
  int k  = (int)(i & 8191);
  wpT[i] = (__bf16)wp[(size_t)k * D_EMB + nn];
}

// ---- kH: h[N,8] = relu(x @ w1) @ w2 (all three matmuls on WMMA) ------------
// 128 WGs x 8 waves; each wave owns 16 rows of N. w1T panels (32 a-cols x
// 1024 k) are async-staged into double-buffered LDS and shared by all waves.
__global__ void __launch_bounds__(256) kH(const __bf16* __restrict__ xb,
                                          const __bf16* __restrict__ w1T,
                                          const __bf16* __restrict__ w2T,
                                          float* __restrict__ h) {
  __shared__ __bf16 Bs[2][32 * BS_STRIDE];   // 2 x 64.5 KB w1T panels
  __shared__ __bf16 R[8][16 * 32];           // per-wave relu tile relayout

  const int tid  = threadIdx.x;
  const int lane = tid & 31;
  const int wv   = tid >> 5;
  const int r    = lane & 15;
  const int hi   = lane >> 4;
  const int m0   = blockIdx.x * 128 + wv * 16;
  __bf16* Rw = &R[wv][0];

  const int NA_TILES = A_ATT / 32;           // 64

  // prologue: start DMA of panel 0
  stage_panel_async(w1T, &Bs[0][0], tid);

  v8f hacc = zero8();                        // 16x16 C tile; cols 0..7 = heads
#pragma unroll 1
  for (int it = 0; it < NA_TILES; ++it) {
    const int a0  = it * 32;
    const int cur = it & 1;
    if (it + 1 < NA_TILES) {
      __syncthreads();  // everyone done reading buffer (it-1) -> safe to fill
      stage_panel_async(w1T + (size_t)(a0 + 32) * D_EMB, &Bs[1 - cur][0], tid);
      asm volatile("s_wait_asynccnt 0x10" ::: "memory");  // panel `it` landed
    } else {
      asm volatile("s_wait_asynccnt 0x0" ::: "memory");
    }
    __syncthreads();    // panel `it` visible to all waves

    const __bf16* Bp = &Bs[cur][0];
    v8f acc0 = zero8();
    v8f acc1 = zero8();
#pragma unroll 4
    for (int k0 = 0; k0 < D_EMB; k0 += 32) {
      v16bf af = frag_ld(xb, D_EMB, m0 + r, k0, hi);       // A from global/L0
      v16bf b0 = frag_ld(Bp, BS_STRIDE, r,      k0, hi);   // B from LDS
      v16bf b1 = frag_ld(Bp, BS_STRIDE, 16 + r, k0, hi);
      acc0 = __builtin_amdgcn_wmma_f32_16x16x32_bf16(false, af, false, b0,
                                                     (short)0, acc0, false, false);
      acc1 = __builtin_amdgcn_wmma_f32_16x16x32_bf16(false, af, false, b1,
                                                     (short)0, acc1, false, false);
    }
    // ReLU + C-layout -> A-layout relayout through wave-private LDS.
    {
      const int n = r;
#pragma unroll
      for (int v = 0; v < 8; ++v) {
        const int m = v + 8 * hi;
        Rw[m * 32 + n]      = (__bf16)fmaxf(acc0[v], 0.0f);
        Rw[m * 32 + 16 + n] = (__bf16)fmaxf(acc1[v], 0.0f);
      }
    }
    asm volatile("s_wait_dscnt 0x0" ::: "memory");  // wave-local LDS RAW
    v16bf a2 = frag_ld(Rw,  32,    r, 0,  hi);      // 16x32 relu tile as A
    v16bf b2 = frag_ld(w2T, A_ATT, r, a0, hi);      // padded w2 slice as B
    hacc = __builtin_amdgcn_wmma_f32_16x16x32_bf16(false, a2, false, b2,
                                                   (short)0, hacc, false, false);
  }
  // store h (cols 0..7 valid)
  const int n = lane & 15;
  if (n < H_HEAD) {
#pragma unroll
    for (int v = 0; v < 8; ++v)
      h[(size_t)(m0 + v + 8 * hi) * H_HEAD + n] = hacc[v];
  }
}

// ---- kS: per-segment softmax, gram, mean(x), x^T@attn ----------------------
__global__ void __launch_bounds__(256) kS(const float* __restrict__ x,
                                          const float* __restrict__ h,
                                          const int* __restrict__ offs,
                                          float* __restrict__ attn_raw,
                                          __bf16* __restrict__ obdh,
                                          float* __restrict__ out_segment,
                                          float* __restrict__ gram_out) {
  __shared__ float red[256][8];
  __shared__ float sgram[64];
  __shared__ float ssum[8];
  __shared__ float smax[8];

  const int b   = blockIdx.x;
  const int t0  = offs[b];
  const int len = offs[b + 1] - t0;
  const int tid = threadIdx.x;

  // phase A: per-head max over the segment
  float mx[8];
#pragma unroll
  for (int j = 0; j < 8; ++j) mx[j] = -3.4e38f;
  for (int t = tid; t < len; t += 256) {
    const float* hp = h + (size_t)(t0 + t) * H_HEAD;
#pragma unroll
    for (int j = 0; j < 8; ++j) mx[j] = fmaxf(mx[j], hp[j]);
  }
#pragma unroll
  for (int j = 0; j < 8; ++j) red[tid][j] = mx[j];
  __syncthreads();
  for (int off = 128; off > 0; off >>= 1) {
    if (tid < off) {
#pragma unroll
      for (int j = 0; j < 8; ++j)
        red[tid][j] = fmaxf(red[tid][j], red[tid + off][j]);
    }
    __syncthreads();
  }
  if (tid < 8)  smax[tid] = red[0][tid];
  if (tid < 64) sgram[tid] = 0.0f;
  if (tid >= 64 && tid < 72) ssum[tid - 64] = 0.0f;
  __syncthreads();

  // phase B: exp, per-head sums, unnormalized gram
  float ls[8];
#pragma unroll
  for (int j = 0; j < 8; ++j) ls[j] = 0.0f;
  float lg[64];
#pragma unroll
  for (int k = 0; k < 64; ++k) lg[k] = 0.0f;
  for (int t = tid; t < len; t += 256) {
    const float* hp = h + (size_t)(t0 + t) * H_HEAD;
    float* ap = attn_raw + (size_t)(t0 + t) * H_HEAD;
    float p[8];
#pragma unroll
    for (int j = 0; j < 8; ++j) {
      p[j] = __expf(hp[j] - smax[j]);
      ap[j] = p[j];
      ls[j] += p[j];
    }
#pragma unroll
    for (int i = 0; i < 8; ++i)
#pragma unroll
      for (int j = 0; j < 8; ++j) lg[i * 8 + j] += p[i] * p[j];
  }
#pragma unroll
  for (int j = 0; j < 8; ++j) atomicAdd(&ssum[j], ls[j]);
#pragma unroll
  for (int k = 0; k < 64; ++k) atomicAdd(&sgram[k], lg[k]);
  __syncthreads();

  if (tid < 64) {
    const int i = tid >> 3, j = tid & 7;
    gram_out[(size_t)b * 64 + tid] = sgram[tid] / (ssum[i] * ssum[j]);
  }

  // phase D: out_segment (mean of x) and out_bdh = x^T @ attn (bf16 for kP)
  float inv_s[8];
#pragma unroll
  for (int j = 0; j < 8; ++j) inv_s[j] = 1.0f / ssum[j];
  const float invlen = 1.0f / (float)len;
  const int d0 = tid * 4;
  float acc[4][8] = {};
  float xs[4] = {};
  for (int t = 0; t < len; ++t) {
    __builtin_prefetch(x + (size_t)(t0 + t + 8) * D_EMB + d0, 0, 1);
    const float4 xv = *reinterpret_cast<const float4*>(
        x + (size_t)(t0 + t) * D_EMB + d0);
    const float4 p0 = *reinterpret_cast<const float4*>(
        attn_raw + (size_t)(t0 + t) * H_HEAD);
    const float4 p1 = *reinterpret_cast<const float4*>(
        attn_raw + (size_t)(t0 + t) * H_HEAD + 4);
    const float pv[8] = {p0.x, p0.y, p0.z, p0.w, p1.x, p1.y, p1.z, p1.w};
    const float xc[4] = {xv.x, xv.y, xv.z, xv.w};
#pragma unroll
    for (int c = 0; c < 4; ++c) {
      xs[c] += xc[c];
#pragma unroll
      for (int j = 0; j < 8; ++j) acc[c][j] += xc[c] * pv[j];
    }
  }
#pragma unroll
  for (int c = 0; c < 4; ++c) {
    out_segment[(size_t)b * D_EMB + d0 + c] = xs[c] * invlen;
#pragma unroll
    for (int j = 0; j < 8; ++j)
      obdh[(size_t)b * (D_EMB * H_HEAD) + (size_t)(d0 + c) * H_HEAD + j] =
          (__bf16)(acc[c][j] * inv_s[j]);
  }
}

// ---- kP: out[64,1024] = obdh[64,8192] @ w_post[8192,1024] on WMMA ----------
__global__ void __launch_bounds__(256) kP(const __bf16* __restrict__ obdh,
                                          const __bf16* __restrict__ wpT,
                                          float* __restrict__ out) {
  const int lane = threadIdx.x & 31;
  const int wv   = blockIdx.x * 8 + (threadIdx.x >> 5);  // 0..255
  const int mt   = wv >> 6;       // 4 row tiles of 16 (M=64)
  const int nt   = wv & 63;       // 64 col tiles of 16 (N=1024)
  const int r    = lane & 15;
  const int hi   = lane >> 4;
  const int K    = D_EMB * H_HEAD;  // 8192

  v8f acc = zero8();
#pragma unroll 4
  for (int k0 = 0; k0 < K; k0 += 32) {
    v16bf af = frag_ld(obdh, K, mt * 16 + r, k0, hi);
    v16bf bf = frag_ld(wpT,  K, nt * 16 + r, k0, hi);
    acc = __builtin_amdgcn_wmma_f32_16x16x32_bf16(false, af, false, bf,
                                                  (short)0, acc, false, false);
  }
  const int n = lane & 15;
#pragma unroll
  for (int v = 0; v < 8; ++v)
    out[(size_t)(mt * 16 + v + 8 * hi) * D_EMB + nt * 16 + n] = acc[v];
}

// ---------------------------------------------------------------------------
extern "C" void kernel_launch(void* const* d_in, const int* in_sizes, int n_in,
                              void* d_out, int out_size, void* d_ws, size_t ws_size,
                              hipStream_t stream) {
  const float* x    = (const float*)d_in[0];
  const int*   seg  = (const int*)d_in[1];
  const float* w1   = (const float*)d_in[2];
  const float* w2   = (const float*)d_in[3];
  const float* wp   = (const float*)d_in[4];

  float* out         = (float*)d_out;                       // [B, D]
  float* out_segment = out + (size_t)B_SEG * D_EMB;         // [B, D]
  float* gram_out    = out_segment + (size_t)B_SEG * D_EMB; // [B, H, H]

  // workspace carve-up (256B aligned)
  char* ws = (char*)d_ws;
  size_t off = 0;
  auto carve = [&](size_t bytes) {
    void* p = ws + off;
    off = (off + bytes + 255) & ~(size_t)255;
    return p;
  };
  __bf16* xb    = (__bf16*)carve((size_t)N_FR * D_EMB * 2);            // 33.5 MB
  __bf16* w1T   = (__bf16*)carve((size_t)A_ATT * D_EMB * 2);           //  4.2 MB
  __bf16* w2T   = (__bf16*)carve((size_t)16 * A_ATT * 2);              //   64 KB
  __bf16* wpT   = (__bf16*)carve((size_t)D_EMB * D_EMB * H_HEAD * 2);  // 16.8 MB
  float*  hbuf  = (float*)carve((size_t)N_FR * H_HEAD * 4);            //  512 KB
  float*  araw  = (float*)carve((size_t)N_FR * H_HEAD * 4);            //  512 KB
  __bf16* obdh  = (__bf16*)carve((size_t)B_SEG * D_EMB * H_HEAD * 2);  //    1 MB
  int*    offs  = (int*)carve((size_t)(B_SEG + 1) * 4);
  (void)ws_size; (void)n_in; (void)in_sizes; (void)out_size;

  k_offs <<<1, 64, 0, stream>>>(seg, offs);
  k_cvt_x<<<(N_FR * D_EMB) / 256, 256, 0, stream>>>(x, xb);
  k_w1T  <<<(A_ATT * D_EMB) / 256, 256, 0, stream>>>(w1, w1T);
  k_w2T  <<<(16 * A_ATT) / 256, 256, 0, stream>>>(w2, w2T);
  k_wpT  <<<(D_EMB * D_EMB * H_HEAD) / 256, 256, 0, stream>>>(wp, wpT);

  kH<<<N_FR / 128, 256, 0, stream>>>(xb, w1T, w2T, hbuf);
  kS<<<B_SEG, 256, 0, stream>>>(x, hbuf, offs, araw, obdh, out_segment, gram_out);
  kP<<<32, 256, 0, stream>>>(obdh, wpT, out);
}
